// FullyConnectedEncoder_29051158790530
// MI455X (gfx1250) — compile-verified, matrix operand order checked by reading, not run on hardware
//
#include <hip/hip_runtime.h>
#include <hip/hip_bf16.h>

// ---------------------------------------------------------------------------
// FullyConnectedEncoder for MI455X (gfx1250, wave32, WMMA).
// Dense per-scene attention (fully-connected 128-node blocks) + WMMA GEMMs.
// Weights are pre-swizzled once per launch into f16 WMMA-fragment order;
// GEMM B tiles then stream through double-buffered LDS via
// GLOBAL_LOAD_ASYNC_TO_LDS_B128 and each lane reads its fragment as one
// contiguous 32-byte LDS access.
// ---------------------------------------------------------------------------

typedef __attribute__((ext_vector_type(16))) _Float16 v16h;
typedef __attribute__((ext_vector_type(8)))  float    v8f;
typedef __attribute__((ext_vector_type(4)))  int      v4i;

#define NN   2048   // total nodes  (B*P*A)
#define DD   128    // embed dim
#define BB   16     // scenes
#define PP   4      // patches
#define AA   32     // agents
#define HH   8      // heads
#define DHH  16     // head dim
#define NMODES 6

// ------------------------- async-copy plumbing -----------------------------
#if __has_builtin(__builtin_amdgcn_global_load_async_to_lds_b128)
#define HAVE_ASYNC_LDS 1
typedef __attribute__((address_space(1))) v4i gas_v4i;   // global 128-bit chunk
typedef __attribute__((address_space(3))) v4i las_v4i;   // LDS 128-bit chunk
#else
#define HAVE_ASYNC_LDS 0
#endif

static __device__ __forceinline__ void async_wait_all() {
#if HAVE_ASYNC_LDS
#if __has_builtin(__builtin_amdgcn_s_wait_asynccnt)
  __builtin_amdgcn_s_wait_asynccnt(0);
#else
  asm volatile("s_wait_asynccnt 0x0" ::: "memory");
#endif
#endif
}

// Stage one pre-swizzled 32x64 weight tile (2048 f16 = 4 KB, contiguous).
static __device__ __forceinline__ void stage_sw(const _Float16* __restrict__ src,
                                                _Float16* __restrict__ dst) {
  for (int c = threadIdx.x; c < 256; c += 128) {    // 256 16-byte chunks
#if HAVE_ASYNC_LDS
    __builtin_amdgcn_global_load_async_to_lds_b128((gas_v4i*)(src + c * 8),
                                                   (las_v4i*)(dst + c * 8), 0, 0);
#else
    *(float4*)(dst + c * 8) = *(const float4*)(src + c * 8);
#endif
  }
}

// ---------------------------------------------------------------------------
// WMMA fragment loaders (ISA 7.12.2 layouts, wave32).
// ---------------------------------------------------------------------------
template <typename T>
static __device__ __forceinline__ v16h load_a16x32(const T* src, int ld, int kmax) {
  const int lane = threadIdx.x & 31;
  const int r  = lane & 15;
  const int hi = lane >> 4;
  v16h f;
#pragma unroll
  for (int j = 0; j < 8; ++j) {
    const int kb = ((j & 4) << 2) + (hi << 3) + ((j & 3) << 1);
    f[2 * j]     = (kb     < kmax) ? (_Float16)src[r * ld + kb]     : (_Float16)0.0f;
    f[2 * j + 1] = (kb + 1 < kmax) ? (_Float16)src[r * ld + kb + 1] : (_Float16)0.0f;
  }
  return f;
}

template <typename T>
static __device__ __forceinline__ v16h load_b32x16(const T* src, int ld, int kmax) {
  const int lane = threadIdx.x & 31;
  const int c  = lane & 15;
  const int hi = lane >> 4;
  v16h f;
#pragma unroll
  for (int j = 0; j < 8; ++j) {
    const int kb = ((j & 4) << 2) + (hi << 3) + ((j & 3) << 1);
    f[2 * j]     = (kb     < kmax) ? (_Float16)src[kb * ld + c]       : (_Float16)0.0f;
    f[2 * j + 1] = (kb + 1 < kmax) ? (_Float16)src[(kb + 1) * ld + c] : (_Float16)0.0f;
  }
  return f;
}

// B operand where the logical B = X^T : B[kb][c] = src[c*ld + kb]
template <typename T>
static __device__ __forceinline__ v16h load_bT32x16(const T* src, int ld, int kmax) {
  const int lane = threadIdx.x & 31;
  const int c  = lane & 15;
  const int hi = lane >> 4;
  v16h f;
#pragma unroll
  for (int j = 0; j < 8; ++j) {
    const int kb = ((j & 4) << 2) + (hi << 3) + ((j & 3) << 1);
    f[2 * j]     = (kb     < kmax) ? (_Float16)src[c * ld + kb]     : (_Float16)0.0f;
    f[2 * j + 1] = (kb + 1 < kmax) ? (_Float16)src[c * ld + kb + 1] : (_Float16)0.0f;
  }
  return f;
}

// ---------------------------------------------------------------------------
// Weight swizzle: f32 row-major W[K,N] -> f16 fragment-ordered Wsw.
// Tile (k0/32, n0/64) stored contiguously as [ct 0..3][lane 0..31][e 0..15],
// inverting the load_b32x16 mapping: element (kb, c) -> lane = c|(hi<<4),
// e = 2j + (kb&1), j = ((kb>>4)<<2) | ((kb>>1)&3), hi = (kb>>3)&1.
// ---------------------------------------------------------------------------
__global__ void swizzle_w_kernel(const float* __restrict__ W,
                                 _Float16* __restrict__ Wsw, int N, int K) {
  const int idx = blockIdx.x * blockDim.x + threadIdx.x;
  if (idx >= N * K) return;
  const int k = idx / N, n = idx - k * N;
  const int kb = k & 31;
  const int c  = n & 15;
  const int ct = (n >> 4) & 3;
  const int hi = (kb >> 3) & 1;
  const int j  = ((kb >> 4) << 2) | ((kb >> 1) & 3);
  const int e  = 2 * j + (kb & 1);
  const int lane = c | (hi << 4);
  const size_t tile = (size_t)(k >> 5) * (N >> 6) + (n >> 6);
  Wsw[tile * 2048 + (ct * 32 + lane) * 16 + e] = (_Float16)W[idx];
}

// ---------------------------------------------------------------------------
// WMMA GEMM body: C[M,N] = act(A[M,K] @ W[K,N] + bias)
// A: f16 row-major (global). Wsw: pre-swizzled f16 fragments, streamed via
// double-buffered async LDS tiles. Block = 128 threads = 4 waves; block tile
// 64x64; wave tile 16x64; K step 32. N % 64 == 0, K % 32 == 0.
// ---------------------------------------------------------------------------
template <bool RELU, typename TOUT>
static __device__ void gemm_body(const _Float16* __restrict__ A,
                                 const _Float16* __restrict__ Wsw,
                                 const float* __restrict__ bias,
                                 TOUT* __restrict__ C,
                                 int N, int K,
                                 _Float16 (*Bt)[2048]) {
  const int wave = threadIdx.x >> 5;
  const int lane = threadIdx.x & 31;
  const int rb   = blockIdx.x * 64 + wave * 16;   // row base for this wave
  const int cb0  = blockIdx.y * 64;               // col base for this block
  const int ntiles = N >> 6;

  v8f acc[4] = {};
  stage_sw(Wsw + (size_t)blockIdx.y * 2048, Bt[0]);
  int ib = 0;
  for (int k0 = 0; k0 < K; k0 += 32, ib ^= 1) {
    async_wait_all();
    __syncthreads();                              // staged tile visible to all
    if (k0 + 32 < K) {                            // overlap next tile copy
      const size_t tile = (size_t)((k0 + 32) >> 5) * ntiles + blockIdx.y;
      stage_sw(Wsw + tile * 2048, Bt[ib ^ 1]);
      __builtin_prefetch(A + rb * K + k0 + 32, 0, 1);
    }
    const v16h a = load_a16x32(A + rb * K + k0, K, 32);
#pragma unroll
    for (int ct = 0; ct < 4; ++ct) {
      const v16h b = *(const v16h*)(Bt[ib] + ct * 512 + lane * 16);
      acc[ct] = __builtin_amdgcn_wmma_f32_16x16x32_f16(
          false, a, false, b, (short)0, acc[ct], false, false);
    }
    __syncthreads();                              // done reading before restage
  }

  const int n  = lane & 15;
  const int hi = lane >> 4;
#pragma unroll
  for (int ct = 0; ct < 4; ++ct) {
    const int c0 = cb0 + ct * 16;
#pragma unroll
    for (int i = 0; i < 8; ++i) {
      const int m = rb + i + hi * 8;       // D layout: VGPR i -> rows i, i+8
      float v = acc[ct][i];
      if (bias) v += bias[c0 + n];
      if (RELU) v = fmaxf(v, 0.0f);
      C[m * N + c0 + n] = (TOUT)v;
    }
  }
}

template <bool RELU, typename TOUT>
__global__ void gemm_wmma(const _Float16* __restrict__ A, const _Float16* __restrict__ Wsw,
                          const float* __restrict__ bias, TOUT* __restrict__ C,
                          int N, int K) {
  __shared__ __align__(32) _Float16 Bt[2][2048];  // 8 KB double buffer
  gemm_body<RELU, TOUT>(A, Wsw, bias, C, N, K, Bt);
}

// Fused q/k/v/s projections: gridDim.z selects the weight set.
struct QKVSArgs { const _Float16* W[4]; const float* B[4]; };
__global__ void qkvs_gemm_kernel(const _Float16* __restrict__ A, QKVSArgs args,
                                 _Float16* __restrict__ out) {
  __shared__ __align__(32) _Float16 Bt[2][2048];
  const int z = blockIdx.z;
  gemm_body<false, _Float16>(A, args.W[z], args.B[z],
                             out + (size_t)z * NN * DD, DD, DD, Bt);
}

// ---------------------------------------------------------------------------
// Dense attention for one (scene, head): q,k,v slices are [128, 16].
// S = (q @ k^T) * DH^-0.5  -> row softmax -> agg = P @ v.
// Scores in 64 KB LDS (f32); probabilities written back in place as f16
// at half-stride 256 so rows never alias across threads.
// ---------------------------------------------------------------------------
__global__ void attn_kernel(const _Float16* __restrict__ q,
                            const _Float16* __restrict__ k,
                            const _Float16* __restrict__ v,
                            float* __restrict__ agg) {
  __shared__ float S[128 * 128];                 // 64 KB (<= 320 KB WGP LDS)
  const int bh  = blockIdx.x;                    // 0..127
  const int b   = bh >> 3;
  const int h   = bh & 7;
  const int nb  = b * 128;                       // node base of this scene
  const int col = h * DHH;                       // head column base
  const int wave = threadIdx.x >> 5;
  const int lane = threadIdx.x & 31;
  const int ln   = lane & 15;
  const int lhi  = lane >> 4;

  // ---- scores: 8x8 grid of 16x16 WMMA tiles, K = 16 padded to 32 ----------
  for (int t = wave; t < 64; t += 4) {
    const int mt = t >> 3, nt = t & 7;
    const v16h a  = load_a16x32 (q + (nb + mt * 16) * DD + col, DD, DHH);
    const v16h bt = load_bT32x16(k + (nb + nt * 16) * DD + col, DD, DHH);
    v8f c = {};
    c = __builtin_amdgcn_wmma_f32_16x16x32_f16(false, a, false, bt,
                                               (short)0, c, false, false);
#pragma unroll
    for (int i = 0; i < 8; ++i)
      S[(mt * 16 + i + lhi * 8) * 128 + nt * 16 + ln] = c[i] * 0.25f; // DH^-0.5
  }
  __syncthreads();

  // ---- per-row softmax: thread r owns row r entirely ----------------------
  {
    const int r = threadIdx.x;
    float* row = S + r * 128;
    float mx = -3.4e38f;
    for (int j = 0; j < 128; ++j) mx = fmaxf(mx, row[j]);
    float sum = 0.0f;
    for (int j = 0; j < 128; ++j) { const float e = __expf(row[j] - mx); row[j] = e; sum += e; }
    const float inv = 1.0f / sum;
    _Float16* prow = (_Float16*)row;             // f16 probs, stride 256 halves
    for (int j = 0; j < 128; ++j) prow[j] = (_Float16)(row[j] * inv);
  }
  __syncthreads();

  // ---- agg = P @ v : 8 row tiles, K = 128 in 4 WMMA steps -----------------
  const _Float16* Ph = (const _Float16*)S;       // row m at Ph + m*256
  for (int mt = wave; mt < 8; mt += 4) {
    v8f c = {};
#pragma unroll
    for (int kc = 0; kc < 4; ++kc) {
      const v16h a  = load_a16x32(Ph + (mt * 16) * 256 + kc * 32, 256, 32);
      const v16h bb = load_b32x16(v + (nb + kc * 32) * DD + col, DD, 32);
      c = __builtin_amdgcn_wmma_f32_16x16x32_f16(false, a, false, bb,
                                                 (short)0, c, false, false);
    }
#pragma unroll
    for (int i = 0; i < 8; ++i)
      agg[(nb + mt * 16 + i + lhi * 8) * DD + col + ln] = c[i];
  }
}

// ---------------------------------------------------------------------------
// Elementwise / reduction kernels
// ---------------------------------------------------------------------------
__global__ void permute_in_kernel(const float* __restrict__ pe, float* __restrict__ x) {
  const int n = blockIdx.x, t = threadIdx.x;
  const int b = n >> 7, p = (n >> 5) & 3, a = n & 31;
  x[n * DD + t] = pe[(p * (BB * AA) + b * AA + a) * DD + t];
}

__global__ void ln_f16_kernel(const float* __restrict__ x, const float* __restrict__ g,
                              const float* __restrict__ bia, _Float16* __restrict__ out) {
  __shared__ float red[DD];
  const int row = blockIdx.x, t = threadIdx.x;
  const float v = x[row * DD + t];
  red[t] = v; __syncthreads();
  for (int s = DD / 2; s > 0; s >>= 1) { if (t < s) red[t] += red[t + s]; __syncthreads(); }
  const float mu = red[0] * (1.0f / DD);
  __syncthreads();
  const float d = v - mu;
  red[t] = d * d; __syncthreads();
  for (int s = DD / 2; s > 0; s >>= 1) { if (t < s) red[t] += red[t + s]; __syncthreads(); }
  const float inv = rsqrtf(red[0] * (1.0f / DD) + 1e-5f);
  out[row * DD + t] = (_Float16)(d * inv * g[t] + bia[t]);
}

__global__ void residual_ln_kernel(float* __restrict__ x, const float* __restrict__ src,
                                   const float* __restrict__ g, const float* __restrict__ bia) {
  __shared__ float red[DD];
  const int row = blockIdx.x, t = threadIdx.x;
  const float v = src[row * DD + t];
  red[t] = v; __syncthreads();
  for (int s = DD / 2; s > 0; s >>= 1) { if (t < s) red[t] += red[t + s]; __syncthreads(); }
  const float mu = red[0] * (1.0f / DD);
  __syncthreads();
  const float d = v - mu;
  red[t] = d * d; __syncthreads();
  for (int s = DD / 2; s > 0; s >>= 1) { if (t < s) red[t] += red[t + s]; __syncthreads(); }
  const float inv = rsqrtf(red[0] * (1.0f / DD) + 1e-5f);
  x[row * DD + t] += d * inv * g[t] + bia[t];
}

__global__ void concat_kernel(const float* __restrict__ agg, const _Float16* __restrict__ h,
                              _Float16* __restrict__ cat) {
  const int idx = blockIdx.x * blockDim.x + threadIdx.x;
  if (idx >= NN * 2 * DD) return;
  const int n = idx / (2 * DD), j = idx - n * 2 * DD;
  cat[idx] = (j < DD) ? (_Float16)agg[n * DD + j] : h[n * DD + (j - DD)];
}

__global__ void gate_kernel(const float* __restrict__ agg, const float* __restrict__ gpre,
                            const _Float16* __restrict__ s, _Float16* __restrict__ gated) {
  const int idx = blockIdx.x * blockDim.x + threadIdx.x;
  if (idx >= NN * DD) return;
  const float gg = 1.0f / (1.0f + __expf(-gpre[idx]));
  const float sv = (float)s[idx];
  gated[idx] = (_Float16)(agg[idx] + gg * (sv - agg[idx]));
}

__global__ void gather_last_kernel(const float* __restrict__ x, _Float16* __restrict__ xa) {
  const int r = blockIdx.x, t = threadIdx.x;
  const int b = r >> 5, a = r & 31;
  const int node = (b * PP + (PP - 1)) * AA + a;
  xa[r * DD + t] = (_Float16)x[node * DD + t];
}

__global__ void out_permute_kernel(const float* __restrict__ y, float* __restrict__ out) {
  const int idx = blockIdx.x * blockDim.x + threadIdx.x;
  if (idx >= NMODES * (BB * AA) * DD) return;
  const int d = idx & (DD - 1);
  const int n = (idx / DD) & (BB * AA - 1);
  const int m = idx / (DD * BB * AA);
  out[idx] = y[n * (NMODES * DD) + m * DD + d];
}

// ---------------------------------------------------------------------------
// Host launch. Input flattening assumed in setup_inputs() dict insertion
// order: patch_embed, params.layers[0..2] (23 leaves each, insertion order),
// params.Wp, params.bp, edge_index (unused: topology is static).
// ---------------------------------------------------------------------------
extern "C" void kernel_launch(void* const* d_in, const int* in_sizes, int n_in,
                              void* d_out, int out_size, void* d_ws, size_t ws_size,
                              hipStream_t stream) {
  (void)in_sizes; (void)n_in; (void)out_size; (void)ws_size;
  const float* patch = (const float*)d_in[0];
  auto LP = [&](int l, int j) { return (const float*)d_in[1 + l * 23 + j]; };
  const float* Wp = (const float*)d_in[1 + 3 * 23 + 0];
  const float* bp = (const float*)d_in[1 + 3 * 23 + 1];

  char* wp = (char*)d_ws;
  auto carve = [&](size_t bytes) {
    char* p = wp; wp += (bytes + 255) & ~(size_t)255; return p;
  };
  float*    x     = (float*)   carve(NN * DD * 4);
  _Float16* h     = (_Float16*)carve(NN * DD * 2);
  _Float16* qkvs  = (_Float16*)carve((size_t)4 * NN * DD * 2);
  float*    aggb  = (float*)   carve(NN * DD * 4);
  _Float16* catb  = (_Float16*)carve(NN * 2 * DD * 2);
  float*    gpre  = (float*)   carve(NN * DD * 4);
  _Float16* gated = (_Float16*)carve(NN * DD * 2);
  float*    tmp   = (float*)   carve(NN * DD * 4);
  _Float16* ff1   = (_Float16*)carve(NN * 4 * DD * 2);
  _Float16* xa    = (_Float16*)carve(BB * AA * DD * 2);
  float*    yb    = (float*)   carve(BB * AA * NMODES * DD * 4);

  // Swizzled f16 weights: per layer q,k,v,s (128x128), Wg (256x128),
  // Wo (128x128), W1 (128x512), W2 (512x128); plus Wp (128x768).
  const int wsz[8] = {DD * DD, DD * DD, DD * DD, DD * DD,
                      2 * DD * DD, DD * DD, DD * 4 * DD, 4 * DD * DD};
  _Float16* wsw[3][8];
  for (int l = 0; l < 3; ++l)
    for (int i = 0; i < 8; ++i)
      wsw[l][i] = (_Float16*)carve((size_t)wsz[i] * 2);
  _Float16* wpsw = (_Float16*)carve((size_t)DD * NMODES * DD * 2);

  // ---- pre-swizzle all weights into WMMA fragment order -------------------
  auto swz = [&](const float* W, _Float16* dst, int K, int N) {
    swizzle_w_kernel<<<(N * K + 255) / 256, 256, 0, stream>>>(W, dst, N, K);
  };
  for (int l = 0; l < 3; ++l) {
    swz(LP(l, 2),  wsw[l][0], DD, DD);        // Wq
    swz(LP(l, 4),  wsw[l][1], DD, DD);        // Wk
    swz(LP(l, 5),  wsw[l][2], DD, DD);        // Wv
    swz(LP(l, 7),  wsw[l][3], DD, DD);        // Ws
    swz(LP(l, 9),  wsw[l][4], 2 * DD, DD);    // Wg
    swz(LP(l, 11), wsw[l][5], DD, DD);        // Wo
    swz(LP(l, 17), wsw[l][6], DD, 4 * DD);    // W1
    swz(LP(l, 19), wsw[l][7], 4 * DD, DD);    // W2
  }
  swz(Wp, wpsw, DD, NMODES * DD);

  _Float16* qf = qkvs;
  _Float16* kf = qkvs + (size_t)1 * NN * DD;
  _Float16* vf = qkvs + (size_t)2 * NN * DD;
  _Float16* sf = qkvs + (size_t)3 * NN * DD;

  permute_in_kernel<<<NN, DD, 0, stream>>>(patch, x);

  const dim3 gProj(NN / 64, DD / 64);            // [2048 x 128] tiles
  const dim3 gQKVS(NN / 64, DD / 64, 4);         // fused 4-way projection
  const dim3 gFF1 (NN / 64, (4 * DD) / 64);      // [2048 x 512]
  for (int l = 0; l < 3; ++l) {
    ln_f16_kernel<<<NN, DD, 0, stream>>>(x, LP(l, 0), LP(l, 1), h);

    QKVSArgs qa;
    qa.W[0] = wsw[l][0]; qa.B[0] = LP(l, 3);     // Wq, bq
    qa.W[1] = wsw[l][1]; qa.B[1] = nullptr;      // Wk (no bias)
    qa.W[2] = wsw[l][2]; qa.B[2] = LP(l, 6);     // Wv, bv
    qa.W[3] = wsw[l][3]; qa.B[3] = LP(l, 8);     // Ws, bs
    qkvs_gemm_kernel<<<gQKVS, 128, 0, stream>>>(h, qa, qkvs);

    attn_kernel<<<BB * HH, 128, 0, stream>>>(qf, kf, vf, aggb);

    concat_kernel<<<(NN * 2 * DD + 255) / 256, 256, 0, stream>>>(aggb, h, catb);
    gemm_wmma<false, float><<<gProj, 128, 0, stream>>>(catb, wsw[l][4], LP(l, 10), gpre, DD, 2 * DD);
    gate_kernel<<<(NN * DD + 255) / 256, 256, 0, stream>>>(aggb, gpre, sf, gated);
    gemm_wmma<false, float><<<gProj, 128, 0, stream>>>(gated, wsw[l][5], LP(l, 12), tmp, DD, DD);
    residual_ln_kernel<<<NN, DD, 0, stream>>>(x, tmp, LP(l, 13), LP(l, 14));

    ln_f16_kernel<<<NN, DD, 0, stream>>>(x, LP(l, 15), LP(l, 16), h);
    gemm_wmma<true, _Float16><<<gFF1, 128, 0, stream>>>(h, wsw[l][6], LP(l, 18), ff1, 4 * DD, DD);
    gemm_wmma<false, float  ><<<gProj, 128, 0, stream>>>(ff1, wsw[l][7], LP(l, 20), tmp, DD, 4 * DD);
    residual_ln_kernel<<<NN, DD, 0, stream>>>(x, tmp, LP(l, 21), LP(l, 22));
  }

  gather_last_kernel<<<BB * AA, DD, 0, stream>>>(x, xa);
  const dim3 gOut((BB * AA) / 64, (NMODES * DD) / 64);
  gemm_wmma<false, float><<<gOut, 128, 0, stream>>>(xa, wpsw, bp, yb, NMODES * DD, DD);
  out_permute_kernel<<<(NMODES * BB * AA * DD + 255) / 256, 256, 0, stream>>>(yb, (float*)d_out);
}
